// KANEXAttention_10831907521099
// MI455X (gfx1250) — matrix operand order repaired and checked by reading, not compiled, not run on hardware
//
#include <hip/hip_runtime.h>
#include <math.h>

typedef __attribute__((ext_vector_type(16))) __bf16 v16bf;
typedef __attribute__((ext_vector_type(8)))  float  v8f;
typedef unsigned int u32x4 __attribute__((ext_vector_type(4)));
typedef int          i32x8 __attribute__((ext_vector_type(8)));
typedef int          i32x4 __attribute__((ext_vector_type(4)));

#define DIM      1024
#define SEQ      2048
#define BATCH    2
#define HEADS    16
#define HEAD_DIM 64
#define WINDOW   64
#define MTOK     (BATCH*SEQ)        // 4096 tokens
#define QKVROW   (3*HEADS*HEAD_DIM) // 3072

// ---------------------------------------------------------------------------
// Fragment layout maps (CDNA5 ISA §7.12.2, bf16 16x16x32)
// A 16x32: lane = (m&15) + 16*((k>>3)&1); element e built from k as below.
// B 32x16: lane = (n&15) + 16*((k>>3)&1); element e = 8*((k>>4)&1) + (k&7).
// ---------------------------------------------------------------------------
__device__ __forceinline__ int a_kmap(int e, int half) {
  int p = e >> 1, sub = e & 1;
  return sub + 2*(p & 3) + 8*half + 16*(p >> 2);
}

// Scatter-store one fp32 value into a packed bf16 A-fragment-ordered buffer.
__device__ __forceinline__ void store_packed_a(__bf16* P, int m, int k, int K, float v) {
  int chunk = k >> 5, kl = k & 31;
  int half  = (kl >> 3) & 1;
  int k2    = kl - 8*half;             // {0..7} U {16..23}
  int hi    = k2 >> 4;
  int e     = 2*(4*hi + ((k2 & 7) >> 1)) + (kl & 1);
  int lane  = (m & 15) + 16*half;
  P[((size_t)((m >> 4)*(K >> 5) + chunk)*32 + lane)*16 + e] = (__bf16)v;
}

// ---------------------------------------------------------------------------
// Pack fp32 row-major [M,K] -> bf16 A-fragment order. 8 k-elems per thread.
// For 8-aligned k0 the 8 destinations are contiguous -> single b128 store.
// ---------------------------------------------------------------------------
__global__ __launch_bounds__(256)
void pack_a_kernel(const float* __restrict__ A, __bf16* __restrict__ P, int M, int K) {
  int idx = blockIdx.x * blockDim.x + threadIdx.x;   // one per 8 elements
  int kq  = idx % (K >> 3);
  int m   = idx / (K >> 3);
  int k0  = kq * 8;
  const float4* src = (const float4*)(A + (size_t)m * K + k0);
  float4 v0 = src[0], v1 = src[1];
  int half = (k0 >> 3) & 1;
  int lane = (m & 15) + 16*half;
  int e0   = ((k0 >> 4) & 1) * 8;
  __bf16* dst = P + ((size_t)((m >> 4)*(K >> 5) + (k0 >> 5))*32 + lane)*16 + e0;
  union { __bf16 h[8]; uint4 u; } t;
  t.h[0]=(__bf16)v0.x; t.h[1]=(__bf16)v0.y; t.h[2]=(__bf16)v0.z; t.h[3]=(__bf16)v0.w;
  t.h[4]=(__bf16)v1.x; t.h[5]=(__bf16)v1.y; t.h[6]=(__bf16)v1.z; t.h[7]=(__bf16)v1.w;
  *(uint4*)dst = t.u;
}

// Pack weights fp32 [N,K] -> bf16 B-fragment order (lane=n&15 + 16*k.bit3, e=8*k.bit4 + (k&7)).
__global__ __launch_bounds__(256)
void pack_w_kernel(const float* __restrict__ W, __bf16* __restrict__ P, int N, int K) {
  int idx = blockIdx.x * blockDim.x + threadIdx.x;
  int kq  = idx % (K >> 3);
  int n   = idx / (K >> 3);
  int k0  = kq * 8;
  const float4* src = (const float4*)(W + (size_t)n * K + k0);
  float4 v0 = src[0], v1 = src[1];
  int lane = (n & 15) + 16*((k0 >> 3) & 1);
  int e0   = ((k0 >> 4) & 1) * 8;                      // 0 or 8
  __bf16* dst = P + ((size_t)((n >> 4)*(K >> 5) + (k0 >> 5))*32 + lane)*16 + e0;
  union { __bf16 h[8]; uint4 u; } t;
  t.h[0]=(__bf16)v0.x; t.h[1]=(__bf16)v0.y; t.h[2]=(__bf16)v0.z; t.h[3]=(__bf16)v0.w;
  t.h[4]=(__bf16)v1.x; t.h[5]=(__bf16)v1.y; t.h[6]=(__bf16)v1.z; t.h[7]=(__bf16)v1.w;
  *(uint4*)dst = t.u;
}

// ---------------------------------------------------------------------------
// TDM: load one B panel stage (4 n-subtiles x 8 k-chunks = 32KB) into LDS.
// 2D tile: 4 rows of 8KB (1024 x 8B units), tensor row stride 32KB (4096 units).
// ---------------------------------------------------------------------------
__device__ __forceinline__ void tdm_load_b_panel(const __bf16* gsrc, unsigned lds_byte_addr) {
  unsigned long long ga = (unsigned long long)(const void*)gsrc;
  u32x4 g0;
  g0[0] = 1u;                                        // count=1, user descriptor
  g0[1] = lds_byte_addr;                             // lds_addr
  g0[2] = (unsigned)(ga & 0xFFFFFFFFu);              // global_addr[31:0]
  g0[3] = (unsigned)((ga >> 32) & 0x1FFFFFFu) | (2u << 30);  // addr[56:32] | type=2
  i32x8 g1;
  g1[0] = (int)(3u << 16);                           // data_size = 8B
  g1[1] = (int)((4096u & 0xFFFFu) << 16);            // tensor_dim0[15:0]=4096
  g1[2] = (int)((4096u >> 16) | (16u << 16));        // tensor_dim0 hi | tensor_dim1=16
  g1[3] = (int)(1024u << 16);                        // tile_dim0 = 1024 units (8KB)
  g1[4] = 4;                                         // tile_dim1 = 4 rows
  g1[5] = 4096;                                      // tensor_dim0_stride = 32KB
  g1[6] = 0;
  g1[7] = 0;
  i32x4 z4 = {0, 0, 0, 0};
#if __clang_major__ >= 23
  i32x8 z8 = {0, 0, 0, 0, 0, 0, 0, 0};
  __builtin_amdgcn_tensor_load_to_lds(g0, g1, z4, z4, z8, 0);
#else
  __builtin_amdgcn_tensor_load_to_lds(g0, g1, z4, z4, 0);
#endif
  __builtin_amdgcn_s_wait_tensorcnt(0);
}

// ---------------------------------------------------------------------------
// Packed-operand GEMM: C = act(A @ W^T + bias).
// A: bf16 A-fragment packed [M/16][K/32][32 lanes][16].
// W: bf16 B-fragment packed [N/16][K/32][32 lanes][16].
// Block 256 thr = 8 waves; tile M=128 (wave=16 rows) x N=64 (4 accumulators).
// B panel staged via TDM in 4 stages of 32KB; inner loop is pure b128 + wmma.
// ---------------------------------------------------------------------------
template<int ACT, bool PACKOUT>
__global__ __launch_bounds__(256)
void gemm_packed_wmma(const __bf16* __restrict__ Ap, const __bf16* __restrict__ Wp,
                      const float* __restrict__ bias,
                      float* __restrict__ Cf, __bf16* __restrict__ Cp,
                      int M, int N, int K) {
  __shared__ __bf16 Bs[16384];                       // 32KB stage buffer
  const int tid   = threadIdx.x;
  const int lane  = tid & 31;
  const int wave  = tid >> 5;
  const int half  = lane >> 4;
  const int l16   = lane & 15;
  const int m0    = blockIdx.y * 128;
  const int n0    = blockIdx.x * 64;
  const int tileM = (m0 >> 4) + wave;
  const int tileN0 = n0 >> 4;
  const int nchunks = K >> 5;

  v8f acc[4] = {};
  for (int s = 0; s < 4; ++s) {
    if (wave == 0)
      tdm_load_b_panel(Wp + (size_t)(tileN0 * nchunks + s * 8) * 512,
                       (unsigned)(size_t)(void*)&Bs[0]);
    __syncthreads();
    // prefetch next stage of this wave's A stream
    if (s < 3)
      __builtin_prefetch(Ap + ((size_t)(tileM * nchunks + (s + 1) * 8) * 32 + lane) * 16, 0, 0);
#pragma unroll
    for (int c8 = 0; c8 < 8; ++c8) {
      const int chunk = s * 8 + c8;
      v16bf a = *(const v16bf*)(Ap + ((size_t)(tileM * nchunks + chunk) * 32 + lane) * 16);
#pragma unroll
      for (int t = 0; t < 4; ++t) {
        v16bf b = *(const v16bf*)&Bs[(t * 8 + c8) * 512 + lane * 16];
        acc[t] = __builtin_amdgcn_wmma_f32_16x16x32_bf16(false, a, false, b,
                                                         (short)0, acc[t], false, false);
      }
    }
    __syncthreads();
  }
  // epilogue: bias + activation, fp32 row-major or packed-bf16 A-order output
#pragma unroll
  for (int t = 0; t < 4; ++t) {
    const int n  = n0 + t * 16 + l16;
    const float bn = bias[n];
#pragma unroll
    for (int r = 0; r < 8; ++r) {
      int m = m0 + wave * 16 + r + 8 * half;         // C layout: row = r + 8*laneHalf
      float v = acc[t][r] + bn;
      if (ACT) v = v / (1.0f + __expf(-v));          // SiLU
      if (PACKOUT) store_packed_a(Cp, m, n, N, v);
      else         Cf[(size_t)m * N + n] = v;
    }
  }
}

// ---------------------------------------------------------------------------
// Flash-style windowed attention. 1 wave per (batch, head, 16-row query tile).
// qkv fp32 [b, n, 3072] = [q|k|v] head-major. Output packed bf16 A-order
// (token index b*SEQ+i, feature k = h*64+d) feeding the final GEMM directly.
// ---------------------------------------------------------------------------
__global__ __launch_bounds__(32)
void attn_wmma(const float* __restrict__ qkv, __bf16* __restrict__ outp) {
  __shared__ __bf16 pS[16][16];
  const int lane = threadIdx.x & 31;
  const int half = lane >> 4;
  const int l16  = lane & 15;
  const int i0 = blockIdx.x * 16;
  const int hh = blockIdx.y;
  const int b  = blockIdx.z;
  const float* base = qkv + (size_t)b * SEQ * QKVROW;
  const int qoff = hh * HEAD_DIM;
  const int koff = HEADS*HEAD_DIM + hh*HEAD_DIM;
  const int voff = 2*HEADS*HEAD_DIM + hh*HEAD_DIM;

  v16bf aq[2];
  {
    const float* qrow = base + (size_t)(i0 + l16) * QKVROW + qoff;
#pragma unroll
    for (int kk = 0; kk < 2; ++kk)
#pragma unroll
      for (int e = 0; e < 16; ++e)
        aq[kk][e] = (__bf16)qrow[a_kmap(e, half) + 32*kk];
  }

  v8f o[4] = {};
  float mrow[8], lrow[8];
#pragma unroll
  for (int e = 0; e < 8; ++e) { mrow[e] = -3.0e38f; lrow[e] = 0.0f; }

  const int jt_max = min((i0 + 15 + WINDOW - 1) >> 4, SEQ/16 - 1);
  for (int jt = 0; jt <= jt_max; ++jt) {
    const int j0 = jt * 16;
    v16bf bk[2];
    {
      const float* krow = base + (size_t)(j0 + l16) * QKVROW + koff;
#pragma unroll
      for (int kk = 0; kk < 2; ++kk) {
        const float* p = krow + 32*kk;
#pragma unroll
        for (int e = 0; e < 16; ++e) bk[kk][e] = (__bf16)p[a_kmap(e, half)];
      }
    }
    v8f s = {};
    s = __builtin_amdgcn_wmma_f32_16x16x32_bf16(false, aq[0], false, bk[0],
                                                (short)0, s, false, false);
    s = __builtin_amdgcn_wmma_f32_16x16x32_bf16(false, aq[1], false, bk[1],
                                                (short)0, s, false, false);

    float p8[8];
#pragma unroll
    for (int e = 0; e < 8; ++e) {
      const int irow = i0 + e + 8*half;
      const int j    = j0 + l16;
      float v = s[e] * 0.125f;                       // 1/sqrt(64)
      if (j >= irow + WINDOW) v = -1.0e30f;
      p8[e] = v;
    }
#pragma unroll
    for (int e = 0; e < 8; ++e) {
      float rmax = p8[e];
#pragma unroll
      for (int off = 1; off < 16; off <<= 1)
        rmax = fmaxf(rmax, __shfl_xor(rmax, off, 32));
      float mnew  = fmaxf(mrow[e], rmax);
      float alpha = __expf(mrow[e] - mnew);
      float pe    = __expf(p8[e] - mnew);
      float rsum  = pe;
#pragma unroll
      for (int off = 1; off < 16; off <<= 1)
        rsum += __shfl_xor(rsum, off, 32);
      lrow[e] = lrow[e] * alpha + rsum;
      mrow[e] = mnew;
      p8[e]   = pe;
#pragma unroll
      for (int t = 0; t < 4; ++t) o[t][e] *= alpha;
    }

    __syncthreads();
#pragma unroll
    for (int e = 0; e < 8; ++e) pS[e + 8*half][l16] = (__bf16)p8[e];
    __syncthreads();
    v16bf ap;
#pragma unroll
    for (int e = 0; e < 16; ++e) {
      int k = 16*half + e;
      ap[e] = (k < 16) ? pS[l16][k] : (__bf16)0.0f;
    }
#pragma unroll
    for (int t = 0; t < 4; ++t) {
      v16bf bv;
      if (half == 0) {
#pragma unroll
        for (int e = 0; e < 16; ++e)
          bv[e] = (__bf16)base[(size_t)(j0 + e)*QKVROW + voff + t*16 + l16];
      } else {
#pragma unroll
        for (int e = 0; e < 16; ++e) bv[e] = (__bf16)0.0f;
      }
      o[t] = __builtin_amdgcn_wmma_f32_16x16x32_bf16(false, ap, false, bv,
                                                     (short)0, o[t], false, false);
    }
    __syncthreads();
  }

  float invl[8];
#pragma unroll
  for (int e = 0; e < 8; ++e) invl[e] = 1.0f / lrow[e];
#pragma unroll
  for (int t = 0; t < 4; ++t)
#pragma unroll
    for (int e = 0; e < 8; ++e) {
      int mtok = b*SEQ + i0 + e + 8*half;
      int k    = hh*HEAD_DIM + t*16 + l16;
      store_packed_a(outp, mtok, k, HEADS*HEAD_DIM, o[t][e] * invl[e]);
    }
}

// ---------------------------------------------------------------------------
extern "C" void kernel_launch(void* const* d_in, const int* in_sizes, int n_in,
                              void* d_out, int out_size, void* d_ws, size_t ws_size,
                              hipStream_t stream) {
  (void)in_sizes; (void)n_in; (void)out_size; (void)ws_size;
  const float* x  = (const float*)d_in[0];
  const float* W1 = (const float*)d_in[1];
  const float* b1 = (const float*)d_in[2];
  const float* W2 = (const float*)d_in[3];
  const float* b2 = (const float*)d_in[4];
  const float* Wg = (const float*)d_in[5];
  const float* bg = (const float*)d_in[6];
  const float* Wo = (const float*)d_in[7];
  const float* bo = (const float*)d_in[8];
  float* y = (float*)d_out;

  char* w = (char*)d_ws;
  __bf16* xa  = (__bf16*)w; w += (size_t)MTOK * DIM * 2;        // 8MB  (reused as attp)
  __bf16* W1p = (__bf16*)w; w += (size_t)DIM * DIM * 2;         // 2MB
  __bf16* W2p = (__bf16*)w; w += (size_t)DIM * DIM * 2;         // 2MB
  __bf16* Wgp = (__bf16*)w; w += (size_t)QKVROW * DIM * 2;      // 6MB
  __bf16* Wop = (__bf16*)w; w += (size_t)DIM * DIM * 2;         // 2MB
  __bf16* tp  = (__bf16*)w; w += (size_t)MTOK * DIM * 2;        // 8MB
  __bf16* hp  = (__bf16*)w; w += (size_t)MTOK * DIM * 2;        // 8MB
  float*  qkv = (float*)w;  w += (size_t)MTOK * QKVROW * 4;     // 48MB
  __bf16* attp = xa;                                            // reuse

  dim3 blk(256);
  // one-time packing (bf16 halves bytes; everything stays L2-resident)
  pack_a_kernel<<<dim3((MTOK*DIM/8)/256),      blk, 0, stream>>>(x,  xa,  MTOK,  DIM);
  pack_w_kernel<<<dim3((DIM*DIM/8)/256),       blk, 0, stream>>>(W1, W1p, DIM,   DIM);
  pack_w_kernel<<<dim3((DIM*DIM/8)/256),       blk, 0, stream>>>(W2, W2p, DIM,   DIM);
  pack_w_kernel<<<dim3((QKVROW*DIM/8)/256),    blk, 0, stream>>>(Wg, Wgp, QKVROW, DIM);
  pack_w_kernel<<<dim3((DIM*DIM/8)/256),       blk, 0, stream>>>(Wo, Wop, DIM,   DIM);

  gemm_packed_wmma<1, true ><<<dim3(DIM/64,    MTOK/128), blk, 0, stream>>>(xa,  W1p, b1, nullptr, tp, MTOK, DIM,    DIM);
  gemm_packed_wmma<0, true ><<<dim3(DIM/64,    MTOK/128), blk, 0, stream>>>(tp,  W2p, b2, nullptr, hp, MTOK, DIM,    DIM);
  gemm_packed_wmma<0, false><<<dim3(QKVROW/64, MTOK/128), blk, 0, stream>>>(hp,  Wgp, bg, qkv, nullptr, MTOK, QKVROW, DIM);
  attn_wmma<<<dim3(SEQ/16, HEADS, BATCH), dim3(32), 0, stream>>>(qkv, attp);
  gemm_packed_wmma<0, false><<<dim3(DIM/64,    MTOK/128), blk, 0, stream>>>(attp, Wop, bo, y, nullptr, MTOK, DIM,    DIM);
}
